// HealthGNN_45526653338172
// MI455X (gfx1250) — matrix operand order repaired and checked by reading, not compiled, not run on hardware
//
#include <hip/hip_runtime.h>
#include <cstdint>
#include <cstddef>

// ---------------------------------------------------------------------------
// Types for CDNA5 WMMA (wave32)
// ---------------------------------------------------------------------------
typedef __attribute__((ext_vector_type(16))) __bf16 v16bf;
typedef __attribute__((ext_vector_type(8)))  __bf16 v8bf;
typedef __attribute__((ext_vector_type(8)))  float  v8f;

// ---------------------------------------------------------------------------
// Elementwise helpers
// ---------------------------------------------------------------------------
__global__ void setval_k(float* __restrict__ p, float v, size_t n) {
    size_t i = (size_t)blockIdx.x * blockDim.x + threadIdx.x;
    if (i < n) p[i] = v;
}

// x [n,Ksrc] f32 -> xbf [n,Kpad] bf16 (zero-padded K)
__global__ void conv_x_k(const float* __restrict__ x, __bf16* __restrict__ xbf,
                         int n, int Ksrc, int Kpad) {
    size_t idx = (size_t)blockIdx.x * blockDim.x + threadIdx.x;
    size_t total = (size_t)n * Kpad;
    if (idx >= total) return;
    int k = (int)(idx % Kpad);
    size_t row = idx / Kpad;
    float v = (k < Ksrc) ? x[row * Ksrc + k] : 0.0f;
    xbf[idx] = (__bf16)v;
}

// Pack W [Ksrc,Nout] f32 (row-major) into CDNA5 B-fragment layout, bf16.
// Packed index = ((kt*ntiles + nt)*32 + lane)*16 + e
// Source K = kt*32 + e + 16*(lane>>4), N = nt*16 + (lane&15). Zero-pad k>=Ksrc.
__global__ void pack_b_k(const float* __restrict__ W, __bf16* __restrict__ Bpk,
                         int Ksrc, int Kpad, int Nout) {
    int idx = blockIdx.x * blockDim.x + threadIdx.x;
    int ntiles = Nout >> 4;
    int total = (Kpad >> 5) * ntiles * 32 * 16;
    if (idx >= total) return;
    int e    = idx & 15;
    int lane = (idx >> 4) & 31;
    int tile = idx >> 9;           // kt*ntiles + nt
    int kt = tile / ntiles, nt = tile % ntiles;
    int k    = kt * 32 + e + 16 * (lane >> 4);
    int ncol = nt * 16 + (lane & 15);
    float v = (k < Ksrc) ? W[(size_t)k * Nout + ncol] : 0.0f;
    Bpk[idx] = (__bf16)v;
}

// ---------------------------------------------------------------------------
// WMMA GEMM: C[M,Nout] f32 = A[M,K] bf16 (row-major) @ Bpk (pre-packed bf16)
// One wave per 16 x (16*NT) tile. M % 16 == 0, K % 32 == 0, EXEC all-ones.
// ---------------------------------------------------------------------------
template <int NT>
__global__ void gemm_wmma_bf16(const __bf16* __restrict__ A,
                               const __bf16* __restrict__ Bpk,
                               float* __restrict__ C,
                               int M, int K, int Nout) {
    const int lane   = threadIdx.x & 31;
    const int row0   = blockIdx.x * 16;
    const int ntbase = blockIdx.y * NT;
    const int ntiles = Nout >> 4;

    v8f acc[NT];
#pragma unroll
    for (int j = 0; j < NT; ++j) acc[j] = (v8f){};

    const int m    = row0 + (lane & 15);
    const int koff = 16 * (lane >> 4);
    const __bf16* arow = A + (size_t)m * K;

    for (int kb = 0; kb < K; kb += 32) {
        v8bf lo = *(const v8bf*)(arow + kb + koff);
        v8bf hi = *(const v8bf*)(arow + kb + 8 + koff);
        v16bf a;
#pragma unroll
        for (int i = 0; i < 8; ++i) { a[i] = lo[i]; a[i + 8] = hi[i]; }
        int kt = kb >> 5;
#pragma unroll
        for (int j = 0; j < NT; ++j) {
            v16bf b = *(const v16bf*)(Bpk +
                (((size_t)kt * ntiles + (ntbase + j)) * 32 + lane) * 16);
            acc[j] = __builtin_amdgcn_wmma_f32_16x16x32_bf16(
                false, a, false, b, (short)0, acc[j], false, false);
        }
    }

    const int mg  = row0 + 8 * (lane >> 4);
    const int col = lane & 15;
#pragma unroll
    for (int j = 0; j < NT; ++j) {
        int cb = (ntbase + j) * 16 + col;
#pragma unroll
        for (int r = 0; r < 8; ++r)
            C[(size_t)(mg + r) * Nout + cb] = acc[j][r];
    }
}

// ---------------------------------------------------------------------------
// Per-node attention coefficients: as[i,h] = <h_pre[i,h,:], a_src[h,:]> etc.
// ---------------------------------------------------------------------------
__global__ void alpha_k(const float* __restrict__ hpre,
                        const float* __restrict__ a_src,
                        const float* __restrict__ a_dst,
                        float* __restrict__ as_, float* __restrict__ ad_,
                        int n, int H, int C) {
    int t = blockIdx.x * blockDim.x + threadIdx.x;
    if (t >= n * H) return;
    int node = t / H, h = t % H;
    const float4* hp  = (const float4*)(hpre + (size_t)node * H * C + (size_t)h * C);
    const float4* av  = (const float4*)(a_src + h * C);
    const float4* bv  = (const float4*)(a_dst + h * C);
    float s1 = 0.f, s2 = 0.f;
    for (int i = 0; i < C / 4; ++i) {
        float4 v = hp[i], a = av[i], b = bv[i];
        s1 += v.x * a.x + v.y * a.y + v.z * a.z + v.w * a.w;
        s2 += v.x * b.x + v.y * b.y + v.z * b.z + v.w * b.w;
    }
    as_[t] = s1;
    ad_[t] = s2;
}

// ---------------------------------------------------------------------------
// Segment softmax over edges (with implicit self-loops appended).
// ---------------------------------------------------------------------------
__device__ __forceinline__ void atomicMaxFloat(float* addr, float val) {
    if (val >= 0.0f) atomicMax((int*)addr, __float_as_int(val));
    else             atomicMin((unsigned int*)addr, __float_as_uint(val));
}

__device__ __forceinline__ void edge_pair(const int* ei, int E, int e, int& s, int& d) {
    if (e < E) { s = ei[e]; d = ei[E + e]; }
    else       { s = e - E; d = e - E; }
}

template <int H>
__global__ void edge_max_k(const int* __restrict__ ei, int E, int N,
                           const float* __restrict__ as_, const float* __restrict__ ad_,
                           float* __restrict__ m) {
    int e = blockIdx.x * blockDim.x + threadIdx.x;
    if (e >= E + N) return;
    int s, d; edge_pair(ei, E, e, s, d);
#pragma unroll
    for (int h = 0; h < H; ++h) {
        float v = as_[s * H + h] + ad_[d * H + h];
        v = v > 0.f ? v : 0.2f * v;                // leaky_relu(0.2)
        atomicMaxFloat(&m[d * H + h], v);
    }
}

template <int H>
__global__ void edge_sum_k(const int* __restrict__ ei, int E, int N,
                           const float* __restrict__ as_, const float* __restrict__ ad_,
                           const float* __restrict__ m, float* __restrict__ denom) {
    int e = blockIdx.x * blockDim.x + threadIdx.x;
    if (e >= E + N) return;
    int s, d; edge_pair(ei, E, e, s, d);
#pragma unroll
    for (int h = 0; h < H; ++h) {
        float v = as_[s * H + h] + ad_[d * H + h];
        v = v > 0.f ? v : 0.2f * v;
        atomicAdd(&denom[d * H + h], expf(v - m[d * H + h]));
    }
}

// Wave-per-edge weighted scatter: out[d,:] += alpha * h_pre[s,:]
template <int H, int C>
__global__ void edge_agg_k(const int* __restrict__ ei, int E, int N,
                           const float* __restrict__ as_, const float* __restrict__ ad_,
                           const float* __restrict__ m, const float* __restrict__ denom,
                           const float* __restrict__ hpre, float* __restrict__ out) {
    const int HC = H * C;
    int wid  = (int)(((size_t)blockIdx.x * blockDim.x + threadIdx.x) >> 5);
    int lane = threadIdx.x & 31;
    if (wid >= E + N) return;
    int s, d; edge_pair(ei, E, wid, s, d);
    float al[H];
#pragma unroll
    for (int h = 0; h < H; ++h) {
        float v = as_[s * H + h] + ad_[d * H + h];
        v = v > 0.f ? v : 0.2f * v;
        al[h] = expf(v - m[d * H + h]) / (denom[d * H + h] + 1e-16f);
    }
    const float* hs = hpre + (size_t)s * HC;
    float* od = out + (size_t)d * HC;
#pragma unroll
    for (int i = 0; i < HC / 32; ++i) {
        int c = lane + 32 * i;
        atomicAdd(&od[c], al[c / C] * hs[c]);
    }
}

// ---------------------------------------------------------------------------
// (agg + bias) -> LayerNorm -> ELU -> bf16 (wave32 per node, D = 256)
// ---------------------------------------------------------------------------
__global__ void ln_elu_k(const float* __restrict__ agg, const float* __restrict__ bias,
                         const float* __restrict__ g, const float* __restrict__ beta,
                         __bf16* __restrict__ out, int n) {
    const int D = 256;
    int node = blockIdx.x * 8 + (threadIdx.x >> 5);
    int lane = threadIdx.x & 31;
    if (node >= n) return;
    float x[8];
    float s = 0.f;
#pragma unroll
    for (int i = 0; i < 8; ++i) {
        int c = lane + 32 * i;
        x[i] = agg[(size_t)node * D + c] + bias[c];
        s += x[i];
    }
#pragma unroll
    for (int off = 16; off > 0; off >>= 1) s += __shfl_xor(s, off, 32);
    float mu = s * (1.0f / 256.0f);
    float vs = 0.f;
#pragma unroll
    for (int i = 0; i < 8; ++i) { float dd = x[i] - mu; vs += dd * dd; }
#pragma unroll
    for (int off = 16; off > 0; off >>= 1) vs += __shfl_xor(vs, off, 32);
    float rs = rsqrtf(vs * (1.0f / 256.0f) + 1e-5f);
#pragma unroll
    for (int i = 0; i < 8; ++i) {
        int c = lane + 32 * i;
        float y = (x[i] - mu) * rs * g[c] + beta[c];
        y = y > 0.f ? y : expm1f(y);               // ELU (alpha=1)
        out[(size_t)node * D + c] = (__bf16)y;
    }
}

// ---------------------------------------------------------------------------
// Final head: h = agg3 + b3 (written back in place to d_out), then
// risk = sigmoid(relu(h@Wr1+br1)@Wr2+br2). Weights staged in LDS.
// ---------------------------------------------------------------------------
__global__ void head_k(const float* __restrict__ b3,
                       const float* __restrict__ Wr1, const float* __restrict__ br1,
                       const float* __restrict__ Wr2, const float* __restrict__ br2,
                       float* __restrict__ out_h, float* __restrict__ out_r, int n) {
    __shared__ float sW1[32 * 32], sW2[32 * 6], sb1[32], sb2[6], sb3[32];
    int tid = threadIdx.x;
    for (int i = tid; i < 1024; i += 256) sW1[i] = Wr1[i];
    if (tid < 192) sW2[tid] = Wr2[tid];
    if (tid < 32)  { sb1[tid] = br1[tid]; sb3[tid] = b3[tid]; }
    if (tid < 6)   sb2[tid] = br2[tid];
    __syncthreads();
    int i = blockIdx.x * 256 + tid;
    if (i >= n) return;
    float h[32];
#pragma unroll
    for (int k = 0; k < 32; ++k) {
        h[k] = out_h[(size_t)i * 32 + k] + sb3[k];
        out_h[(size_t)i * 32 + k] = h[k];
    }
    float r[32];
#pragma unroll
    for (int j = 0; j < 32; ++j) {
        float acc = sb1[j];
#pragma unroll
        for (int k = 0; k < 32; ++k) acc += h[k] * sW1[k * 32 + j];
        r[j] = acc > 0.f ? acc : 0.f;
    }
#pragma unroll
    for (int q = 0; q < 6; ++q) {
        float acc = sb2[q];
#pragma unroll
        for (int j = 0; j < 32; ++j) acc += r[j] * sW2[j * 6 + q];
        out_r[(size_t)i * 6 + q] = 1.0f / (1.0f + expf(-acc));
    }
}

// ---------------------------------------------------------------------------
// Host orchestration
// ---------------------------------------------------------------------------
static inline int cdiv(long long a, long long b) { return (int)((a + b - 1) / b); }

extern "C" void kernel_launch(void* const* d_in, const int* in_sizes, int n_in,
                              void* d_out, int out_size, void* d_ws, size_t ws_size,
                              hipStream_t stream) {
    const float* x      = (const float*)d_in[0];
    const int*   ei     = (const int*)  d_in[1];
    const float* W1     = (const float*)d_in[2];
    const float* a1s    = (const float*)d_in[3];
    const float* a1d    = (const float*)d_in[4];
    const float* b1     = (const float*)d_in[5];
    const float* W2     = (const float*)d_in[6];
    const float* a2s    = (const float*)d_in[7];
    const float* a2d    = (const float*)d_in[8];
    const float* b2     = (const float*)d_in[9];
    const float* W3     = (const float*)d_in[10];
    const float* a3s    = (const float*)d_in[11];
    const float* a3d    = (const float*)d_in[12];
    const float* b3     = (const float*)d_in[13];
    const float* g1     = (const float*)d_in[14];
    const float* beta1  = (const float*)d_in[15];
    const float* g2     = (const float*)d_in[16];
    const float* beta2  = (const float*)d_in[17];
    const float* Wr1    = (const float*)d_in[18];
    const float* br1    = (const float*)d_in[19];
    const float* Wr2    = (const float*)d_in[20];
    const float* br2    = (const float*)d_in[21];

    const int N = in_sizes[0] / 16;   // 50000 (divisible by 16)
    const int E = in_sizes[1] / 2;    // 800000
    const int Etot = E + N;           // + self-loops

    // ---- workspace carve ----
    char* w = (char*)d_ws;
    auto carve = [&](size_t bytes) -> void* {
        void* p = (void*)w;
        w += (bytes + 255) & ~(size_t)255;
        return p;
    };
    __bf16* Xbf  = (__bf16*)carve((size_t)N * 32 * 2);
    __bf16* W1pk = (__bf16*)carve((size_t)32 * 256 * 2);
    __bf16* W2pk = (__bf16*)carve((size_t)256 * 256 * 2);
    __bf16* W3pk = (__bf16*)carve((size_t)256 * 32 * 2);
    float*  hpre = (float*) carve((size_t)N * 256 * 4);
    float*  agg  = (float*) carve((size_t)N * 256 * 4);
    __bf16* hln  = (__bf16*)carve((size_t)N * 256 * 2);
    float*  as_  = (float*) carve((size_t)N * 4 * 4);
    float*  ad_  = (float*) carve((size_t)N * 4 * 4);
    float*  mbuf = (float*) carve((size_t)N * 4 * 4);
    float*  dbuf = (float*) carve((size_t)N * 4 * 4);
    float*  h3   = (float*) carve((size_t)N * 32 * 4);

    float* out_h = (float*)d_out;
    float* out_r = out_h + (size_t)N * 32;

    const float NEG_INF = -__builtin_inff();
    const dim3 blk256(256), blk32(32);

    // ---- weight packing / input conversion (cheap, every call: deterministic) ----
    conv_x_k<<<cdiv((long long)N * 32, 256), blk256, 0, stream>>>(x, Xbf, N, 16, 32);
    pack_b_k<<<cdiv(32 * 256, 256),  blk256, 0, stream>>>(W1, W1pk, 16, 32, 256);
    pack_b_k<<<cdiv(256 * 256, 256), blk256, 0, stream>>>(W2, W2pk, 256, 256, 256);
    pack_b_k<<<cdiv(256 * 32, 256),  blk256, 0, stream>>>(W3, W3pk, 256, 256, 32);

    // ================= Layer 1: 16 -> 4x64 =================
    gemm_wmma_bf16<4><<<dim3(N / 16, 4), blk32, 0, stream>>>(Xbf, W1pk, hpre, N, 32, 256);
    alpha_k<<<cdiv((long long)N * 4, 256), blk256, 0, stream>>>(hpre, a1s, a1d, as_, ad_, N, 4, 64);
    setval_k<<<cdiv((long long)N * 4, 256), blk256, 0, stream>>>(mbuf, NEG_INF, (size_t)N * 4);
    setval_k<<<cdiv((long long)N * 4, 256), blk256, 0, stream>>>(dbuf, 0.f, (size_t)N * 4);
    setval_k<<<cdiv((long long)N * 256, 256), blk256, 0, stream>>>(agg, 0.f, (size_t)N * 256);
    edge_max_k<4><<<cdiv(Etot, 256), blk256, 0, stream>>>(ei, E, N, as_, ad_, mbuf);
    edge_sum_k<4><<<cdiv(Etot, 256), blk256, 0, stream>>>(ei, E, N, as_, ad_, mbuf, dbuf);
    edge_agg_k<4, 64><<<cdiv((long long)Etot * 32, 256), blk256, 0, stream>>>(
        ei, E, N, as_, ad_, mbuf, dbuf, hpre, agg);
    ln_elu_k<<<cdiv(N, 8), blk256, 0, stream>>>(agg, b1, g1, beta1, hln, N);

    // ================= Layer 2: 256 -> 4x64 =================
    gemm_wmma_bf16<4><<<dim3(N / 16, 4), blk32, 0, stream>>>(hln, W2pk, hpre, N, 256, 256);
    alpha_k<<<cdiv((long long)N * 4, 256), blk256, 0, stream>>>(hpre, a2s, a2d, as_, ad_, N, 4, 64);
    setval_k<<<cdiv((long long)N * 4, 256), blk256, 0, stream>>>(mbuf, NEG_INF, (size_t)N * 4);
    setval_k<<<cdiv((long long)N * 4, 256), blk256, 0, stream>>>(dbuf, 0.f, (size_t)N * 4);
    setval_k<<<cdiv((long long)N * 256, 256), blk256, 0, stream>>>(agg, 0.f, (size_t)N * 256);
    edge_max_k<4><<<cdiv(Etot, 256), blk256, 0, stream>>>(ei, E, N, as_, ad_, mbuf);
    edge_sum_k<4><<<cdiv(Etot, 256), blk256, 0, stream>>>(ei, E, N, as_, ad_, mbuf, dbuf);
    edge_agg_k<4, 64><<<cdiv((long long)Etot * 32, 256), blk256, 0, stream>>>(
        ei, E, N, as_, ad_, mbuf, dbuf, hpre, agg);
    ln_elu_k<<<cdiv(N, 8), blk256, 0, stream>>>(agg, b2, g2, beta2, hln, N);

    // ================= Layer 3: 256 -> 32 (1 head) =================
    gemm_wmma_bf16<2><<<dim3(N / 16, 1), blk32, 0, stream>>>(hln, W3pk, h3, N, 256, 32);
    alpha_k<<<cdiv(N, 256), blk256, 0, stream>>>(h3, a3s, a3d, as_, ad_, N, 1, 32);
    setval_k<<<cdiv(N, 256), blk256, 0, stream>>>(mbuf, NEG_INF, (size_t)N);
    setval_k<<<cdiv(N, 256), blk256, 0, stream>>>(dbuf, 0.f, (size_t)N);
    setval_k<<<cdiv((long long)N * 32, 256), blk256, 0, stream>>>(out_h, 0.f, (size_t)N * 32);
    edge_max_k<1><<<cdiv(Etot, 256), blk256, 0, stream>>>(ei, E, N, as_, ad_, mbuf);
    edge_sum_k<1><<<cdiv(Etot, 256), blk256, 0, stream>>>(ei, E, N, as_, ad_, mbuf, dbuf);
    edge_agg_k<1, 32><<<cdiv((long long)Etot * 32, 256), blk256, 0, stream>>>(
        ei, E, N, as_, ad_, mbuf, dbuf, h3, out_h);

    // ================= Risk head =================
    head_k<<<cdiv(N, 256), blk256, 0, stream>>>(b3, Wr1, br1, Wr2, br2, out_h, out_r, N);
}
